// MultiHeadAttention_63591285785296
// MI455X (gfx1250) — matrix operand order repaired and checked by reading, not compiled, not run on hardware
//
#include <hip/hip_runtime.h>

// ---------------------------------------------------------------------------
// MultiHeadAttention forward for gfx1250 (CDNA5, wave32, WMMA f16 -> f32 acc)
// B=2, S=2048, D=1024, H=16, dk=64.
// Reference reshapes WITHOUT transpose, so head h of Q/K/V/ctx is a CONTIGUOUS
// [2048,64] row-major chunk at element offset bh*131072.
// Double-buffered GLOBAL_LOAD_ASYNC_TO_LDS_B128 tile staging (ASYNCcnt +
// s_wait_asynccnt) so tile loads overlap WMMA compute.
// ---------------------------------------------------------------------------

typedef __attribute__((ext_vector_type(16))) _Float16 v16h;  // A/B fragment (8 VGPRs)
typedef __attribute__((ext_vector_type(8)))  _Float16 h8v;   // 16-byte f16 vector
typedef __attribute__((ext_vector_type(4)))  _Float16 h4v;   // 8-byte f16 vector
typedef __attribute__((ext_vector_type(8)))  float    v8f;   // C/D fragment (8 VGPRs)
typedef int i4v __attribute__((vector_size(16)));            // int32x4 (builtin pointee)

#define WMMA_F16(a, b, c) \
  __builtin_amdgcn_wmma_f32_16x16x32_f16(false, (a), false, (b), (short)0, (c), false, false)

// ---- CDNA5 async global->LDS path -----------------------------------------
#if defined(__AMDGCN__) && __has_builtin(__builtin_amdgcn_global_load_async_to_lds_b128)
#define HAVE_ASYNC_BUILTIN 1
#else
#define HAVE_ASYNC_BUILTIN 0
#endif
#if defined(__AMDGCN__) && !HAVE_ASYNC_BUILTIN
#define HAVE_ASYNC_ASM 1
#else
#define HAVE_ASYNC_ASM 0
#endif
#define HAVE_ASYNC (HAVE_ASYNC_BUILTIN || HAVE_ASYNC_ASM)

// per-lane copy of 16 bytes global -> LDS (async when available)
__device__ __forceinline__ void copy16(const _Float16* g, _Float16* l) {
#if HAVE_ASYNC_BUILTIN
  // Builtin signature (from clang diagnostic): (int4 AS1*, int4 AS3*, Ii, Ii).
  // C++ refuses direct addrspace pointer casts, so go through integers:
  // AS1 address == generic 64-bit address; AS3 (32-bit ptr) == addr[31:0].
  typedef __attribute__((address_space(1))) i4v* gp_t;
  typedef __attribute__((address_space(3))) i4v* lp_t;
  __builtin_amdgcn_global_load_async_to_lds_b128(
      (gp_t)(unsigned long long)(const void*)g,
      (lp_t)(unsigned)(unsigned long long)(void*)l, 0, 0);
#elif HAVE_ASYNC_ASM
  // VDST = per-lane LDS byte address (addr[31:0] of generic pointer),
  // VADDR = per-lane 64-bit global address (GV mode).
  unsigned lofs = (unsigned)(unsigned long long)(void*)l;
  asm volatile("global_load_async_to_lds_b128 %0, %1, off"
               :: "v"(lofs), "v"((unsigned long long)(const void*)g)
               : "memory");
#else
  *(h8v*)l = *(const h8v*)g;
#endif
}

__device__ __forceinline__ void wait_async0() {
#if HAVE_ASYNC
#if __has_builtin(__builtin_amdgcn_s_wait_asynccnt)
  __builtin_amdgcn_s_wait_asynccnt(0);
#else
  asm volatile("s_wait_asynccnt 0" ::: "memory");
#endif
#endif
}

// ---- fragment loaders: contiguous K order, same for A and B --------------------
// A-matrix 16x32 f16: lanes 0-15 row M=lane, VGPR j -> K = half*16 + 2j{,+1}.
// base = row-major tile, stride ld.
__device__ __forceinline__ v16h load_a_frag(const _Float16* base, int ld, int lane) {
  const int hf = (lane >> 4) & 1;
  const _Float16* row = base + (lane & 15) * ld;
  v16h a;
#pragma unroll
  for (int j = 0; j < 8; ++j) {
    const int kb = (hf << 4) + 2 * j;
    a[2 * j]     = row[kb];
    a[2 * j + 1] = row[kb + 1];
  }
  return a;
}

// B-matrix 32x16 f16 from a row-major [K][N] tile (stride ld):
// lanes 0-15 cover K=0..15, lanes 16-31 K=16..31; VGPR j -> K = half*16 + 2j{,+1}
__device__ __forceinline__ v16h load_b_frag(const _Float16* base, int ld, int lane) {
  const int hf = (lane >> 4) & 1;
  const int n  = lane & 15;
  v16h b;
#pragma unroll
  for (int j = 0; j < 8; ++j) {
    const int kk = (hf << 4) + 2 * j;
    b[2 * j]     = base[(size_t)kk * ld + n];
    b[2 * j + 1] = base[(size_t)(kk + 1) * ld + n];
  }
  return b;
}

// B-matrix 32x16 where B(kk,n) = tile[n][kk] (B = tile^T, tile row-major,
// stride ld). Adjacent kk are contiguous -> dword LDS reads.
__device__ __forceinline__ v16h load_bt_frag(const _Float16* base, int ld, int lane) {
  const int hf = (lane >> 4) & 1;
  const _Float16* row = base + (lane & 15) * ld;
  v16h b;
#pragma unroll
  for (int j = 0; j < 8; ++j) {
    const int kk = (hf << 4) + 2 * j;
    b[2 * j]     = row[kk];
    b[2 * j + 1] = row[kk + 1];
  }
  return b;
}

// ---------------------------------------------------------------------------
// fp32 -> f16 conversion (vectorized, n multiple of 4)
// ---------------------------------------------------------------------------
__global__ void cvt_f32_f16(const float* __restrict__ in, _Float16* __restrict__ out, int n4) {
  int i = blockIdx.x * blockDim.x + threadIdx.x;
  if (i < n4) {
    float4 v = ((const float4*)in)[i];
    h4v o;
    o[0] = (_Float16)v.x; o[1] = (_Float16)v.y;
    o[2] = (_Float16)v.z; o[3] = (_Float16)v.w;
    ((h4v*)out)[i] = o;
  }
}

// ---------------------------------------------------------------------------
// Tiled WMMA GEMM:  Out[M,N] = A[M,K] * Bw[K,N] + bias[N]
// Block tile 128x128, K-step 32, 256 threads = 8 waves (4x2 wave grid),
// each wave computes a 32x64 sub-tile = 2x4 C fragments.
// Double-buffered LDS tiles with async global->LDS staging.
// ---------------------------------------------------------------------------
#define GT_M 128
#define GT_N 128
#define GT_K 32
#define AS_LD (GT_K + 16)   // 48 halves = 96B rows (16B-aligned)
#define BS_LD (GT_N + 16)   // 144 halves = 288B rows

__device__ __forceinline__ void gemm_issue(
    const _Float16* __restrict__ A, const _Float16* __restrict__ Bw,
    int N, int K, int m_blk, int n_blk, int k0,
    _Float16 (*As)[AS_LD], _Float16 (*Bs)[BS_LD], int tid) {
  // A tile: 128x32 halves = 512 chunks of 8 halves (16B)
#pragma unroll
  for (int c = tid; c < 512; c += 256) {
    int r = c >> 2, col = (c & 3) * 8;
    copy16(A + (size_t)(m_blk + r) * K + k0 + col, &As[r][col]);
  }
  // B tile: 32x128 halves = 512 chunks of 8 halves
#pragma unroll
  for (int c = tid; c < 512; c += 256) {
    int r = c >> 4, col = (c & 15) * 8;
    copy16(Bw + (size_t)(k0 + r) * N + n_blk + col, &Bs[r][col]);
  }
}

template <bool OUT_F32>
__global__ __launch_bounds__(256) void gemm_f16_kernel(
    const _Float16* __restrict__ A, const _Float16* __restrict__ Bw,
    const float* __restrict__ bias, void* __restrict__ Out,
    int M, int N, int K) {
  __shared__ _Float16 As[2][GT_M][AS_LD];
  __shared__ _Float16 Bs[2][GT_K][BS_LD];

  const int tid  = threadIdx.x;
  const int lane = tid & 31;
  const int w    = tid >> 5;   // 0..7
  const int wr   = w >> 1;     // 0..3  -> M offset wr*32
  const int wc   = w & 1;      // 0..1  -> N offset wc*64
  const int m_blk = blockIdx.y * GT_M;
  const int n_blk = blockIdx.x * GT_N;

  v8f acc[2][4];
#pragma unroll
  for (int i = 0; i < 2; ++i)
#pragma unroll
    for (int j = 0; j < 4; ++j) acc[i][j] = (v8f)0.0f;

  // prologue: stage first K-tile into buffer 0
  gemm_issue(A, Bw, N, K, m_blk, n_blk, 0, As[0], Bs[0], tid);
  int cur = 0;

  for (int k0 = 0; k0 < K; k0 += GT_K) {
    wait_async0();       // my async chunks of As[cur]/Bs[cur] are in LDS
    __syncthreads();     // everyone's chunks are in; prev buffer reads retired
    if (k0 + GT_K < K)
      gemm_issue(A, Bw, N, K, m_blk, n_blk, k0 + GT_K, As[cur ^ 1], Bs[cur ^ 1], tid);

    v16h a0 = load_a_frag(&As[cur][wr * 32][0], AS_LD, lane);
    v16h a1 = load_a_frag(&As[cur][wr * 32 + 16][0], AS_LD, lane);
#pragma unroll
    for (int nt = 0; nt < 4; ++nt) {
      v16h b = load_b_frag(&Bs[cur][0][wc * 64 + nt * 16], BS_LD, lane);
      acc[0][nt] = WMMA_F16(a0, b, acc[0][nt]);
      acc[1][nt] = WMMA_F16(a1, b, acc[1][nt]);
    }
    cur ^= 1;
  }

  const int hf = lane >> 4, nl = lane & 15;
#pragma unroll
  for (int mt = 0; mt < 2; ++mt)
#pragma unroll
    for (int nt = 0; nt < 4; ++nt) {
      const int col = n_blk + wc * 64 + nt * 16 + nl;
      const float bv = bias[col];
#pragma unroll
      for (int v = 0; v < 8; ++v) {
        const int row = m_blk + wr * 32 + mt * 16 + hf * 8 + v;
        const float val = acc[mt][nt][v] + bv;
        if (OUT_F32)
          ((float*)Out)[(size_t)row * N + col] = val;
        else
          ((_Float16*)Out)[(size_t)row * N + col] = (_Float16)val;
      }
    }
}

// ---------------------------------------------------------------------------
// Flash attention per contiguous head chunk [2048,64] (f16 in, f16 ctx out).
// grid = 32 heads * 16 query-blocks; block = 256 threads (8 waves).
// Each wave owns a 16-row query strip; online softmax over 16 KV tiles of 128.
// K/V tiles double-buffered with async global->LDS staging.
// ---------------------------------------------------------------------------
#define ATT_S   2048
#define ATT_DK  64
#define KV_LD   (ATT_DK + 16)   // 80 halves = 160B rows
#define P_LD    (128 + 16)      // 144 halves = 288B rows

__device__ __forceinline__ void attn_issue(
    const _Float16* __restrict__ K, const _Float16* __restrict__ V, int kt,
    _Float16 (*Kl)[KV_LD], _Float16 (*Vl)[KV_LD], int tid) {
#pragma unroll
  for (int c = tid; c < 1024; c += 256) {
    int r = c >> 3, col = (c & 7) * 8;
    const size_t g = (size_t)(kt * 128 + r) * ATT_DK + col;
    copy16(K + g, &Kl[r][col]);
    copy16(V + g, &Vl[r][col]);
  }
}

__global__ __launch_bounds__(256) void attn_kernel(
    const _Float16* __restrict__ Qb, const _Float16* __restrict__ Kb,
    const _Float16* __restrict__ Vb, _Float16* __restrict__ Ctx) {
  __shared__ _Float16 Klds[2][128][KV_LD];
  __shared__ _Float16 Vlds[2][128][KV_LD];
  __shared__ _Float16 Plds[8][16][P_LD];

  const int bh = blockIdx.x >> 4;     // 0..31 (b*16 + h)
  const int qb = blockIdx.x & 15;     // query block of 128 rows
  const size_t off = (size_t)bh * ATT_S * ATT_DK;
  const _Float16* Q = Qb + off;
  const _Float16* K = Kb + off;
  const _Float16* V = Vb + off;
  _Float16* O = Ctx + off;

  const int tid = threadIdx.x, lane = tid & 31, w = tid >> 5;
  const int hf = lane >> 4, nl = lane & 15;

  // stage first KV tile while we set up Q
  attn_issue(K, V, 0, Klds[0], Vlds[0], tid);
  int cur = 0;

  // Persistent Q A-fragments for this wave's 16 rows, pre-scaled by 1/sqrt(dk)
  // (1/8 is a power of two -> exact in f16).
  v16h aq[2];
  {
    const _Float16* qbase = Q + (size_t)(qb * 128 + w * 16) * ATT_DK;
#pragma unroll
    for (int kc = 0; kc < 2; ++kc) {
      aq[kc] = load_a_frag(qbase + kc * 32, ATT_DK, lane);
#pragma unroll
      for (int e = 0; e < 16; ++e) aq[kc][e] = aq[kc][e] * (_Float16)0.125f;
    }
  }

  v8f o_acc[4];
#pragma unroll
  for (int i = 0; i < 4; ++i) o_acc[i] = (v8f)0.0f;
  float m_i[8], l_i[8];
#pragma unroll
  for (int v = 0; v < 8; ++v) { m_i[v] = -1e30f; l_i[v] = 0.0f; }

  for (int kt = 0; kt < ATT_S / 128; ++kt) {
    wait_async0();
    __syncthreads();
    if (kt + 1 < ATT_S / 128)
      attn_issue(K, V, kt + 1, Klds[cur ^ 1], Vlds[cur ^ 1], tid);

    // scores strip 16x128: S = (Q/8) * K^T
    v8f sc[8];
#pragma unroll
    for (int nt = 0; nt < 8; ++nt) {
      v16h b0 = load_bt_frag(&Klds[cur][nt * 16][0], KV_LD, lane);
      v16h b1 = load_bt_frag(&Klds[cur][nt * 16][32], KV_LD, lane);
      sc[nt] = WMMA_F16(aq[0], b0, (v8f)0.0f);
      sc[nt] = WMMA_F16(aq[1], b1, sc[nt]);
    }

    // online softmax per row (row = hf*8 + v; 16 lanes of a half share rows)
#pragma unroll
    for (int v = 0; v < 8; ++v) {
      float tmax = sc[0][v];
#pragma unroll
      for (int nt = 1; nt < 8; ++nt) tmax = fmaxf(tmax, sc[nt][v]);
#pragma unroll
      for (int s = 8; s >= 1; s >>= 1) tmax = fmaxf(tmax, __shfl_xor(tmax, s, 32));
      const float mnew = fmaxf(m_i[v], tmax);
      const float corr = __expf(m_i[v] - mnew);
      m_i[v] = mnew;
      float rs = 0.0f;
#pragma unroll
      for (int nt = 0; nt < 8; ++nt) {
        const float p = __expf(sc[nt][v] - mnew);
        sc[nt][v] = p;
        rs += p;
      }
#pragma unroll
      for (int s = 8; s >= 1; s >>= 1) rs += __shfl_xor(rs, s, 32);
      l_i[v] = l_i[v] * corr + rs;
#pragma unroll
      for (int nt2 = 0; nt2 < 4; ++nt2) o_acc[nt2][v] *= corr;
    }

    // stage P (16x128 f16) through this wave's private LDS strip
    // (DS ops are in-order within a wave: no barrier needed)
#pragma unroll
    for (int nt = 0; nt < 8; ++nt)
#pragma unroll
      for (int v = 0; v < 8; ++v)
        Plds[w][hf * 8 + v][nt * 16 + nl] = (_Float16)sc[nt][v];

    // ctx += P * V   (A = P strip, B = V tile rows)
#pragma unroll
    for (int kc = 0; kc < 4; ++kc) {
      v16h ap = load_a_frag(&Plds[w][0][kc * 32], P_LD, lane);
#pragma unroll
      for (int nt2 = 0; nt2 < 4; ++nt2) {
        v16h bv = load_b_frag(&Vlds[cur][kc * 32][nt2 * 16], KV_LD, lane);
        o_acc[nt2] = WMMA_F16(ap, bv, o_acc[nt2]);
      }
    }
    cur ^= 1;
  }

  // normalize and store ctx (f16, contiguous head-chunk layout)
#pragma unroll
  for (int nt2 = 0; nt2 < 4; ++nt2)
#pragma unroll
    for (int v = 0; v < 8; ++v) {
      const int row = qb * 128 + w * 16 + hf * 8 + v;
      const float val = o_acc[nt2][v] / l_i[v];
      O[(size_t)row * ATT_DK + nt2 * 16 + nl] = (_Float16)val;
    }
}

// ---------------------------------------------------------------------------
// Host launcher
// ---------------------------------------------------------------------------
extern "C" void kernel_launch(void* const* d_in, const int* in_sizes, int n_in,
                              void* d_out, int out_size, void* d_ws, size_t ws_size,
                              hipStream_t stream) {
  (void)in_sizes; (void)n_in; (void)out_size; (void)ws_size;
  const int M = 4096;   // B*S
  const int D = 1024;

  const float* x  = (const float*)d_in[0];
  const float* Wq = (const float*)d_in[1];
  const float* bq = (const float*)d_in[2];
  const float* Wk = (const float*)d_in[3];
  const float* bk = (const float*)d_in[4];
  const float* Wv = (const float*)d_in[5];
  const float* bv = (const float*)d_in[6];
  const float* Wo = (const float*)d_in[7];
  const float* bo = (const float*)d_in[8];
  float* out = (float*)d_out;

  // workspace layout (f16): x | Wq | Wk | Wv | Wo | Q | K | V | ctx  (~48 MB)
  const size_t MD = (size_t)M * D;
  const size_t DD = (size_t)D * D;
  _Float16* p   = (_Float16*)d_ws;
  _Float16* xh  = p;  p += MD;
  _Float16* Wqh = p;  p += DD;
  _Float16* Wkh = p;  p += DD;
  _Float16* Wvh = p;  p += DD;
  _Float16* Woh = p;  p += DD;
  _Float16* Qh  = p;  p += MD;
  _Float16* Kh  = p;  p += MD;
  _Float16* Vh  = p;  p += MD;
  _Float16* Ch  = p;  p += MD;

  // 1) fp32 -> f16 conversions
  {
    int n4 = (int)(MD / 4);
    cvt_f32_f16<<<(n4 + 255) / 256, 256, 0, stream>>>(x, xh, n4);
    n4 = (int)(DD / 4);
    cvt_f32_f16<<<(n4 + 255) / 256, 256, 0, stream>>>(Wq, Wqh, n4);
    cvt_f32_f16<<<(n4 + 255) / 256, 256, 0, stream>>>(Wk, Wkh, n4);
    cvt_f32_f16<<<(n4 + 255) / 256, 256, 0, stream>>>(Wv, Wvh, n4);
    cvt_f32_f16<<<(n4 + 255) / 256, 256, 0, stream>>>(Wo, Woh, n4);
  }

  // 2) Q/K/V projections (f16 out, f32 accumulate)
  dim3 ggrid(D / GT_N, M / GT_M);   // (8, 32)
  gemm_f16_kernel<false><<<ggrid, 256, 0, stream>>>(xh, Wqh, bq, (void*)Qh, M, D, D);
  gemm_f16_kernel<false><<<ggrid, 256, 0, stream>>>(xh, Wkh, bk, (void*)Kh, M, D, D);
  gemm_f16_kernel<false><<<ggrid, 256, 0, stream>>>(xh, Wvh, bv, (void*)Vh, M, D, D);

  // 3) flash attention over 32 contiguous head chunks
  attn_kernel<<<32 * (ATT_S / 128), 256, 0, stream>>>(Qh, Kh, Vh, Ch);

  // 4) output projection (f32 out)
  gemm_f16_kernel<true><<<ggrid, 256, 0, stream>>>(Ch, Woh, bo, (void*)out, M, D, D);
}
